// dce_loss_89008902242835
// MI455X (gfx1250) — matrix operand order repaired and checked by reading, not compiled, not run on hardware
//
#include <hip/hip_runtime.h>
#include <hip/hip_bf16.h>
#include <stdint.h>

typedef __bf16 v16bf __attribute__((ext_vector_type(16)));
typedef __bf16 v4bf  __attribute__((ext_vector_type(4)));
typedef float  v8f   __attribute__((ext_vector_type(8)));
typedef unsigned int u32x4 __attribute__((ext_vector_type(4)));
typedef int          i32x8 __attribute__((ext_vector_type(8)));
typedef int          i32x4 __attribute__((ext_vector_type(4)));

#define BDIM 1024   // D (feature dim)
#define CDIM 2048   // C (num centers)
#define MROW 8192   // B (batch)
#define KC   32     // K chunk (bf16 wmma K)
#define LDA  40     // padded LDS row stride in bf16 elems (80B, 16B aligned)
#define BM   128
#define BN   128
#define NTHR 512    // 16 waves: 4 (M) x 4 (N), each wave 32x32

static __device__ __forceinline__ v8f wmma_bf16(v16bf a, v16bf b, v8f c) {
    return __builtin_amdgcn_wmma_f32_16x16x32_bf16(false, a, false, b, (short)0, c, false, false);
}

static __device__ __forceinline__ void split_bf16(float x, __bf16& hi, __bf16& lo) {
    hi = (__bf16)x;
    lo = (__bf16)(x - (float)hi);
}

// A fragment (16x32 bf16): lane l<16 -> row M=l,    K = {0..7, 16..23}
//                          lane l>=16 -> row M=l-16, K = {8..15, 24..31}
static __device__ __forceinline__ v16bf load_a_frag(const __bf16* base, int lane) {
    const int m  = lane & 15;
    const int kb = (lane >> 4) << 3;
    const __bf16* p = base + m * LDA + kb;
    v16bf r;
#pragma unroll
    for (int i = 0; i < 8; ++i) r[i] = p[i];
#pragma unroll
    for (int i = 0; i < 8; ++i) r[8 + i] = p[16 + i];
    return r;
}

// B fragment (32x16 bf16): lane l<16 -> col N=l, K = {0..7, 16..23} ; lane l>=16 -> col N=l-16, K = {8..15, 24..31}
static __device__ __forceinline__ v16bf load_b_frag(const __bf16* base, int lane) {
    const int n  = lane & 15;
    const int kb = (lane >> 4) << 3;
    const __bf16* p = base + n * LDA + kb;
    v16bf r;
#pragma unroll
    for (int i = 0; i < 8; ++i) r[i] = p[i];
#pragma unroll
    for (int i = 0; i < 8; ++i) r[8 + i] = p[16 + i];
    return r;
}

// ---------------- passthrough copy of centers & sigma into d_out head ----------------
__global__ void copy_pass_kernel(const float* __restrict__ centers,
                                 const float* __restrict__ sigma,
                                 float* __restrict__ out) {
    const int i = blockIdx.x * blockDim.x + threadIdx.x;
    const int n4 = (CDIM * BDIM) / 4;
    if (i < n4) {
        const float4* c4 = (const float4*)centers;
        const float4* s4 = (const float4*)sigma;
        float4* o4 = (float4*)out;
        o4[i]      = c4[i];
        o4[n4 + i] = s4[i];
    }
}

// ---------------- const[i] = sum_d sigma[i,d] * centers[i,d]^2 ----------------
__global__ void const_kernel(const float* __restrict__ centers,
                             const float* __restrict__ sigma,
                             float* __restrict__ cst) {
    __shared__ float red[256];
    const int i = blockIdx.x;
    const size_t base = (size_t)i * BDIM;
    float s = 0.f;
    for (int d = threadIdx.x; d < BDIM; d += 256) {
        const float c = centers[base + d];
        s += sigma[base + d] * c * c;
    }
    red[threadIdx.x] = s;
    __syncthreads();
#pragma unroll
    for (int off = 128; off > 0; off >>= 1) {
        if (threadIdx.x < off) red[threadIdx.x] += red[threadIdx.x + off];
        __syncthreads();
    }
    if (threadIdx.x == 0) cst[i] = red[0];
}

// ================= TDM path: pre-split bf16 planes in workspace =================
// A planes: [4][MROW][BDIM] bf16 = {x^2 hi, x^2 lo, x hi, x lo}
// B planes: [4][CDIM][BDIM] bf16 = {sig hi, sig lo, -2sc hi, -2sc lo}
#define APLANE ((size_t)MROW * BDIM)
#define BPLANE ((size_t)CDIM * BDIM)

__global__ void prep_a_kernel(const float* __restrict__ x, __bf16* __restrict__ a4) {
    const size_t i = (size_t)blockIdx.x * blockDim.x + threadIdx.x;   // float4 index
    const float4 v = ((const float4*)x)[i];
    const float vv[4] = {v.x, v.y, v.z, v.w};
    v4bf p1h, p1l, p2h, p2l;
#pragma unroll
    for (int c = 0; c < 4; ++c) {
        __bf16 h, l;
        split_bf16(vv[c] * vv[c], h, l); p1h[c] = h; p1l[c] = l;
        split_bf16(vv[c],         h, l); p2h[c] = h; p2l[c] = l;
    }
    ((v4bf*)(a4 + 0 * APLANE))[i] = p1h;
    ((v4bf*)(a4 + 1 * APLANE))[i] = p1l;
    ((v4bf*)(a4 + 2 * APLANE))[i] = p2h;
    ((v4bf*)(a4 + 3 * APLANE))[i] = p2l;
}

__global__ void prep_b_kernel(const float* __restrict__ centers, const float* __restrict__ sigma,
                              __bf16* __restrict__ b4) {
    const size_t i = (size_t)blockIdx.x * blockDim.x + threadIdx.x;   // float4 index
    const float4 sg = ((const float4*)sigma)[i];
    const float4 ce = ((const float4*)centers)[i];
    const float ss[4] = {sg.x, sg.y, sg.z, sg.w};
    const float cc[4] = {ce.x, ce.y, ce.z, ce.w};
    v4bf p1h, p1l, p2h, p2l;
#pragma unroll
    for (int c = 0; c < 4; ++c) {
        __bf16 h, l;
        split_bf16(ss[c],                 h, l); p1h[c] = h; p1l[c] = l;
        split_bf16(-2.0f * ss[c] * cc[c], h, l); p2h[c] = h; p2l[c] = l;
    }
    ((v4bf*)(b4 + 0 * BPLANE))[i] = p1h;
    ((v4bf*)(b4 + 1 * BPLANE))[i] = p1l;
    ((v4bf*)(b4 + 2 * BPLANE))[i] = p2h;
    ((v4bf*)(b4 + 3 * BPLANE))[i] = p2l;
}

// Issue one TDM load: 3D tile [4 planes][128 rows][64 B] -> LDS, with 16 B pad per 64 B
// (produces the LDA=40 padded layout). rows_total / plane_stride_dw select A vs B tensor.
static __device__ __forceinline__ void tdm_load_tile(uint32_t lds_addr, const __bf16* tile_gptr,
                                                     uint32_t rows_total, uint32_t plane_stride_dw) {
    const uint64_t ga = (uint64_t)(uintptr_t)tile_gptr;
    u32x4 g0; i32x8 g1; i32x4 g2, g3; i32x8 gx;
    g0[0] = 1u;                                   // count=1, is_restore=0, gather off
    g0[1] = lds_addr;                             // LDS byte address
    g0[2] = (uint32_t)ga;                         // global_addr[31:0]
    g0[3] = (uint32_t)((ga >> 32) & 0x1FFFFFFu) | (2u << 30);   // global_addr[56:32], type=2
    g1[0] = (int)((2u << 16) | (1u << 20) | (3u << 22) | (3u << 25));
    //        data_size=4B      pad_enable   pad_int=16dw  pad_amt=4dw ; wg_mask=0
    g1[1] = (int)((512u & 0xFFFFu) << 16);        // abar=0 | tensor_dim0(512 dw) lo16
    g1[2] = (int)((512u >> 16) | ((rows_total & 0xFFFFu) << 16));   // dim0 hi | dim1 lo16
    g1[3] = (int)((rows_total >> 16) | (16u << 16));                // dim1 hi | tile_dim0=16 dw
    g1[4] = (int)(128u | (4u << 16));             // tile_dim1=128 rows | tile_dim2=4 planes
    g1[5] = (int)512u;                            // tensor_dim0_stride lo32 (512 dw row stride)
    g1[6] = (int)((plane_stride_dw & 0xFFFFu) << 16);  // d0_stride hi16=0 | d1_stride lo16
    g1[7] = (int)(plane_stride_dw >> 16);         // d1_stride hi32
    g2[0] = 4;                                    // tensor_dim2 = 4 planes
    g2[1] = 1;                                    // tensor_dim3 (unused)
    g2[2] = 0;                                    // tensor_dim2_stride lo (unused, no 4th dim)
    g2[3] = 0;                                    // stride hi | tile_dim3=0
    g3[0] = 0; g3[1] = 0; g3[2] = 0; g3[3] = 0;
    gx[0] = 0; gx[1] = 0; gx[2] = 0; gx[3] = 0;   // aux block (6-arg toolchain): zero-filled
    gx[4] = 0; gx[5] = 0; gx[6] = 0; gx[7] = 0;
    __builtin_amdgcn_tensor_load_to_lds(g0, g1, g2, g3, gx, 0);
}

// ---------------- shared compute core ----------------
static __device__ __forceinline__ void compute_chunk(const __bf16* __restrict__ sAbuf,
                                                     const __bf16* __restrict__ sBbuf,
                                                     v8f (&acc)[2][2],
                                                     int wave_m, int wave_n, int lane) {
    const int PL = BM * LDA;   // plane stride in LDS elems (same for A and B: BM==BN)
#pragma unroll
    for (int mi = 0; mi < 2; ++mi) {
        const int mrow = (wave_m * 32 + mi * 16) * LDA;
        const v16bf a1h = load_a_frag(sAbuf + 0 * PL + mrow, lane);
        const v16bf a1l = load_a_frag(sAbuf + 1 * PL + mrow, lane);
        const v16bf a2h = load_a_frag(sAbuf + 2 * PL + mrow, lane);
        const v16bf a2l = load_a_frag(sAbuf + 3 * PL + mrow, lane);
#pragma unroll
        for (int ni = 0; ni < 2; ++ni) {
            const int nrow = (wave_n * 32 + ni * 16) * LDA;
            const v16bf b1h = load_b_frag(sBbuf + 0 * PL + nrow, lane);
            const v16bf b1l = load_b_frag(sBbuf + 1 * PL + nrow, lane);
            const v16bf b2h = load_b_frag(sBbuf + 2 * PL + nrow, lane);
            const v16bf b2l = load_b_frag(sBbuf + 3 * PL + nrow, lane);
            v8f c = acc[mi][ni];
            c = wmma_bf16(a1h, b1h, c);
            c = wmma_bf16(a1h, b1l, c);
            c = wmma_bf16(a1l, b1h, c);
            c = wmma_bf16(a2h, b2h, c);
            c = wmma_bf16(a2h, b2l, c);
            c = wmma_bf16(a2l, b2h, c);
            acc[mi][ni] = c;
        }
    }
}

static __device__ __forceinline__ void epilogue(const v8f (&acc)[2][2], const float* __restrict__ cst,
                                                float* __restrict__ out, int blockM, int blockN,
                                                int wave_m, int wave_n, int lane) {
    const int mo  = (lane >> 4) << 3;
    const int col = lane & 15;
#pragma unroll
    for (int mi = 0; mi < 2; ++mi) {
#pragma unroll
        for (int ni = 0; ni < 2; ++ni) {
            const int rowbase = blockM + wave_m * 32 + mi * 16 + mo;
            const int colg    = blockN + wave_n * 32 + ni * 16 + col;
            const float cs = cst[colg];
#pragma unroll
            for (int r = 0; r < 8; ++r) {
                out[(size_t)(rowbase + r) * CDIM + colg] = -(acc[mi][ni][r] + cs);
            }
        }
    }
}

// ---------------- TDM GEMM: LDS staged entirely by the Tensor Data Mover ----------------
__global__ __launch_bounds__(NTHR) void dce_gemm_tdm_kernel(const __bf16* __restrict__ a4,
                                                            const __bf16* __restrict__ b4,
                                                            const float* __restrict__ cst,
                                                            float* __restrict__ out) {
    __shared__ alignas(64) __bf16 sA[2][4 * BM * LDA];
    __shared__ alignas(64) __bf16 sB[2][4 * BN * LDA];

    const int tid    = threadIdx.x;
    const int lane   = tid & 31;
    const int wid    = tid >> 5;
    const int wave_m = wid & 3;
    const int wave_n = wid >> 2;
    const int blockM = blockIdx.x * BM;
    const int blockN = blockIdx.y * BN;

    const __bf16* aTile = a4 + (size_t)blockM * BDIM;   // + kk per chunk
    const __bf16* bTile = b4 + (size_t)blockN * BDIM;

    v8f acc[2][2] = {};

    // prologue: DMA chunk 0 into buffer 0 (wave 0 issues; TENSORcnt tracks)
    if (wid == 0) {
        tdm_load_tile((uint32_t)(uintptr_t)&sA[0][0], aTile + 0, MROW, APLANE / 2);
        tdm_load_tile((uint32_t)(uintptr_t)&sB[0][0], bTile + 0, CDIM, BPLANE / 2);
        __builtin_amdgcn_s_wait_tensorcnt(0);
    }
    __syncthreads();

    const int nchunks = BDIM / KC;
    for (int kc = 0; kc < nchunks; ++kc) {
        const int cur = kc & 1;
        const int nxt = cur ^ 1;
        const bool more = (kc + 1) < nchunks;

        if (more && wid == 0) {
            const int kk = (kc + 1) * KC;
            tdm_load_tile((uint32_t)(uintptr_t)&sA[nxt][0], aTile + kk, MROW, APLANE / 2);
            tdm_load_tile((uint32_t)(uintptr_t)&sB[nxt][0], bTile + kk, CDIM, BPLANE / 2);
        }

        compute_chunk(sA[cur], sB[cur], acc, wave_m, wave_n, lane);

        if (more && wid == 0) __builtin_amdgcn_s_wait_tensorcnt(0);
        __syncthreads();
    }

    epilogue(acc, cst, out, blockM, blockN, wave_m, wave_n, lane);
}

// ---------------- fallback fused GEMM (no big workspace needed) ----------------
static __device__ __forceinline__ void stageA_load(const float* __restrict__ x,
                                                   int blockM, int kk, int tid, float4 (&r)[2]) {
#pragma unroll
    for (int j = 0; j < 2; ++j) {
        const int slot = tid + j * NTHR;
        const int row  = slot >> 3;
        const int kq   = (slot & 7) << 2;
        r[j] = *(const float4*)(x + (size_t)(blockM + row) * BDIM + kk + kq);
    }
}
static __device__ __forceinline__ void stageA_store(__bf16* __restrict__ buf, int tid, const float4 (&r)[2]) {
#pragma unroll
    for (int j = 0; j < 2; ++j) {
        const int slot = tid + j * NTHR;
        const int row  = slot >> 3;
        const int kq   = (slot & 7) << 2;
        const float vv[4] = {r[j].x, r[j].y, r[j].z, r[j].w};
#pragma unroll
        for (int c = 0; c < 4; ++c) {
            const int idx = row * LDA + kq + c;
            __bf16 h, l;
            split_bf16(vv[c] * vv[c], h, l); buf[0 * BM * LDA + idx] = h; buf[1 * BM * LDA + idx] = l;
            split_bf16(vv[c],         h, l); buf[2 * BM * LDA + idx] = h; buf[3 * BM * LDA + idx] = l;
        }
    }
}
static __device__ __forceinline__ void stageB_load(const float* __restrict__ centers,
                                                   const float* __restrict__ sigma,
                                                   int blockN, int kk, int tid,
                                                   float4 (&rs)[2], float4 (&rc)[2]) {
#pragma unroll
    for (int j = 0; j < 2; ++j) {
        const int slot = tid + j * NTHR;
        const int row  = slot >> 3;
        const int kq   = (slot & 7) << 2;
        const size_t g = (size_t)(blockN + row) * BDIM + kk + kq;
        rs[j] = *(const float4*)(sigma   + g);
        rc[j] = *(const float4*)(centers + g);
    }
}
static __device__ __forceinline__ void stageB_store(__bf16* __restrict__ buf, int tid,
                                                    const float4 (&rs)[2], const float4 (&rc)[2]) {
#pragma unroll
    for (int j = 0; j < 2; ++j) {
        const int slot = tid + j * NTHR;
        const int row  = slot >> 3;
        const int kq   = (slot & 7) << 2;
        const float ss[4] = {rs[j].x, rs[j].y, rs[j].z, rs[j].w};
        const float cc[4] = {rc[j].x, rc[j].y, rc[j].z, rc[j].w};
#pragma unroll
        for (int c = 0; c < 4; ++c) {
            const int idx = row * LDA + kq + c;
            __bf16 h, l;
            split_bf16(ss[c],                 h, l); buf[0 * BN * LDA + idx] = h; buf[1 * BN * LDA + idx] = l;
            split_bf16(-2.0f * ss[c] * cc[c], h, l); buf[2 * BN * LDA + idx] = h; buf[3 * BN * LDA + idx] = l;
        }
    }
}

__global__ __launch_bounds__(NTHR) void dce_gemm_fused_kernel(const float* __restrict__ x,
                                                              const float* __restrict__ centers,
                                                              const float* __restrict__ sigma,
                                                              const float* __restrict__ cst,
                                                              float* __restrict__ out) {
    __shared__ alignas(64) __bf16 sA[2][4 * BM * LDA];
    __shared__ alignas(64) __bf16 sB[2][4 * BN * LDA];

    const int tid    = threadIdx.x;
    const int lane   = tid & 31;
    const int wid    = tid >> 5;
    const int wave_m = wid & 3;
    const int wave_n = wid >> 2;
    const int blockM = blockIdx.x * BM;
    const int blockN = blockIdx.y * BN;

    v8f acc[2][2] = {};
    float4 ra[2], rbs[2], rbc[2];

    stageA_load(x, blockM, 0, tid, ra);
    stageB_load(centers, sigma, blockN, 0, tid, rbs, rbc);
    stageA_store(sA[0], tid, ra);
    stageB_store(sB[0], tid, rbs, rbc);
    __syncthreads();

    const int nchunks = BDIM / KC;
    for (int kc = 0; kc < nchunks; ++kc) {
        const int cur = kc & 1;
        const int nxt = cur ^ 1;
        const bool more = (kc + 1) < nchunks;

        if (more) {
            stageA_load(x, blockM, (kc + 1) * KC, tid, ra);
            stageB_load(centers, sigma, blockN, (kc + 1) * KC, tid, rbs, rbc);
        }
        compute_chunk(sA[cur], sB[cur], acc, wave_m, wave_n, lane);
        if (more) {
            stageA_store(sA[nxt], tid, ra);
            stageB_store(sB[nxt], tid, rbs, rbc);
        }
        __syncthreads();
    }

    epilogue(acc, cst, out, blockM, blockN, wave_m, wave_n, lane);
}

extern "C" void kernel_launch(void* const* d_in, const int* in_sizes, int n_in,
                              void* d_out, int out_size, void* d_ws, size_t ws_size,
                              hipStream_t stream) {
    const float* x       = (const float*)d_in[0];
    const float* centers = (const float*)d_in[1];
    const float* sigma   = (const float*)d_in[2];
    float* out = (float*)d_out;

    // workspace layout: [cst: 8KB][pad to 64KB][A planes: 64MB][B planes: 16MB]
    float*  cst = (float*)d_ws;
    __bf16* a4  = (__bf16*)((char*)d_ws + 65536);
    __bf16* b4  = a4 + 4 * APLANE;
    const size_t need = 65536 + 4 * APLANE * 2 + 4 * BPLANE * 2;

    float* out_dist = out + 2 * (size_t)CDIM * BDIM;

    const int n4 = (CDIM * BDIM) / 4;
    copy_pass_kernel<<<(n4 + 255) / 256, 256, 0, stream>>>(centers, sigma, out);
    const_kernel<<<CDIM, 256, 0, stream>>>(centers, sigma, cst);

    dim3 grid(MROW / BM, CDIM / BN);
    if (ws_size >= need) {
        prep_a_kernel<<<(MROW * BDIM / 4) / 256, 256, 0, stream>>>(x, a4);
        prep_b_kernel<<<(CDIM * BDIM / 4) / 256, 256, 0, stream>>>(centers, sigma, b4);
        dce_gemm_tdm_kernel<<<grid, NTHR, 0, stream>>>(a4, b4, cst, out_dist);
    } else {
        dce_gemm_fused_kernel<<<grid, NTHR, 0, stream>>>(x, centers, sigma, cst, out_dist);
    }
}
